// selfAttention_54511724921231
// MI455X (gfx1250) — matrix operand order repaired and checked by reading, not compiled, not run on hardware
//
#include <hip/hip_runtime.h>
#include <hip/hip_bf16.h>

typedef __bf16 bf16_t;
typedef __attribute__((ext_vector_type(16))) __bf16 v16bf;
typedef __attribute__((ext_vector_type(8)))  __bf16 v8bf;
typedef __attribute__((ext_vector_type(8)))  float  v8f;
typedef int v4i_vs __attribute__((vector_size(16)));  // matches async builtin param type
typedef unsigned int u32x4 __attribute__((ext_vector_type(4)));
typedef int i32x4 __attribute__((ext_vector_type(4)));
typedef int i32x8 __attribute__((ext_vector_type(8)));

#define BB 8
#define CC 512
#define LL 2048
#define TT 64

// CDNA5 async global->LDS copies (ASYNCcnt path).
#if defined(__has_builtin)
#if __has_builtin(__builtin_amdgcn_global_load_async_to_lds_b128) && \
    __has_builtin(__builtin_amdgcn_s_wait_asynccnt)
#define USE_ASYNC_LDS 1
#endif
#if __has_builtin(__builtin_amdgcn_tensor_load_to_lds) && \
    __has_builtin(__builtin_amdgcn_s_wait_tensorcnt)
#define USE_TDM 1
#endif
#endif
#ifndef USE_ASYNC_LDS
#define USE_ASYNC_LDS 0
#endif
#ifndef USE_TDM
#define USE_TDM 0
#endif
// amdgpu-toolchain (clang-23 + therock headers) uses the 6-arg TDM builtin and is
// the toolchain that ships the gfx1250 TDM header; ROCm 7.2 uses the 5-arg form.
#if defined(__has_include)
#if __has_include(<hip/amd_detail/amd_gfx1250_TDM.h>)
#define TDM_SIX_ARG 1
#endif
#endif
#ifndef TDM_SIX_ARG
#define TDM_SIX_ARG 0
#endif

static __device__ __forceinline__ v8f wmma_bf16(v16bf a, v16bf b, v8f c) {
  // D = A(16x32 bf16) x B(32x16 bf16) + C(16x16 f32)
  return __builtin_amdgcn_wmma_f32_16x16x32_bf16(false, a, false, b, (short)0, c,
                                                 false, false);
}

// A-fragment (16x32, bf16, row-major, stride in elements).
// Contiguous k order used here (not ISA table order; B loaders match): lane<16 -> M=lane, K in {half*16+0..15}
static __device__ __forceinline__ v16bf load_a_bf16g(const bf16_t* base, int stride, int lane) {
  const int m = lane & 15, half = lane >> 4;
  const bf16_t* p = base + (size_t)m * stride + half * 16;
  v8bf lo = *(const v8bf*)(p);
  v8bf hi = *(const v8bf*)(p + 8);
  v16bf a;
#pragma unroll
  for (int e = 0; e < 8; ++e) { a[e] = lo[e]; a[e + 8] = hi[e]; }
  return a;
}

// A-fragment from f32 memory with convert to bf16.
static __device__ __forceinline__ v16bf load_a_f32g(const float* base, int stride, int lane) {
  const int m = lane & 15, half = lane >> 4;
  const float* p = base + (size_t)m * stride + half * 16;
  v16bf a;
#pragma unroll
  for (int e = 0; e < 8; ++e) { a[e] = (__bf16)p[e]; a[e + 8] = (__bf16)p[8 + e]; }
  return a;
}

// B-fragment (32x16, bf16). Memory layout is [n][k] (k contiguous).
// Contiguous k order used here (same as A loaders): lane<16 -> N=lane, K=0..15 ; lane>=16 -> N=lane-16, K=16..31.
static __device__ __forceinline__ v16bf load_b_bf16g(const bf16_t* base, int stride, int lane) {
  const int n = lane & 15, kh = lane >> 4;
  const bf16_t* p = base + (size_t)n * stride + kh * 16;
  v8bf lo = *(const v8bf*)(p);
  v8bf hi = *(const v8bf*)(p + 8);
  v16bf b;
#pragma unroll
  for (int e = 0; e < 8; ++e) { b[e] = lo[e]; b[e + 8] = hi[e]; }
  return b;
}

// B-fragment from LDS staging tile sX[32][128] (k-major), columns colbase..+15.
static __device__ __forceinline__ v16bf load_b_lds_bf16(const bf16_t* sX, int colbase, int lane) {
  const int n = lane & 15, kh = lane >> 4;
  v16bf b;
#pragma unroll
  for (int e = 0; e < 16; ++e) b[e] = sX[(kh * 16 + e) * 128 + colbase + n];
  return b;
}

// B-fragment from f32 LDS score strip sS[32][LL]; convert f32->bf16 on the fly.
static __device__ __forceinline__ v16bf load_b_lds_f32(const float* sS, int nbase, int kb, int lane) {
  const int n = lane & 15, kh = lane >> 4;
  const float* p = sS + (size_t)(nbase + n) * LL + kb + kh * 16;
  float4 f0 = *(const float4*)(p);
  float4 f1 = *(const float4*)(p + 4);
  float4 f2 = *(const float4*)(p + 8);
  float4 f3 = *(const float4*)(p + 12);
  v16bf b;
  b[0]  = (__bf16)f0.x; b[1]  = (__bf16)f0.y; b[2]  = (__bf16)f0.z; b[3]  = (__bf16)f0.w;
  b[4]  = (__bf16)f1.x; b[5]  = (__bf16)f1.y; b[6]  = (__bf16)f1.z; b[7]  = (__bf16)f1.w;
  b[8]  = (__bf16)f2.x; b[9]  = (__bf16)f2.y; b[10] = (__bf16)f2.z; b[11] = (__bf16)f2.w;
  b[12] = (__bf16)f3.x; b[13] = (__bf16)f3.y; b[14] = (__bf16)f3.z; b[15] = (__bf16)f3.w;
  return b;
}

// ---------------------------------------------------------------------------
// Kernel 0: x (f32) -> xbf (bf16)
// ---------------------------------------------------------------------------
__global__ void xcvt_kernel(const float* __restrict__ x, bf16_t* __restrict__ xb, int n) {
  int i = blockIdx.x * blockDim.x + threadIdx.x;
  const int stride = gridDim.x * blockDim.x;
  for (; i < n; i += stride) xb[i] = (__bf16)x[i];
}

// ---------------------------------------------------------------------------
// Kernel 1: q = W1 x X, k = W2 x X per batch; store transposed [b][l][t] bf16.
// Block = 256 threads (8 waves), one batch, 128 columns of L.
// GEMM: M=T=64 (4 tiles), N=128 (1 tile per wave), K=C=512 (16 steps of 32).
// X K-block staged to LDS via TDM (TENSOR_LOAD_TO_LDS) when available, else
// CDNA5 async global->LDS, else plain load/ds_store.
// ---------------------------------------------------------------------------
__global__ void __launch_bounds__(256) qk_kernel(const float* __restrict__ W1,
                                                 const float* __restrict__ W2,
                                                 const bf16_t* __restrict__ xbf,
                                                 bf16_t* __restrict__ qT,
                                                 bf16_t* __restrict__ kT) {
  __shared__ bf16_t sX[32 * 128];  // K-block of X, k-major (8 KB), LDS offset 0
  const int b = blockIdx.x >> 4;
  const int lbase = (blockIdx.x & 15) * 128;
  const int tid = threadIdx.x;
  const int lane = tid & 31, wave = tid >> 5;
  const int row = tid >> 3;   // 0..31 : k row of staging tile
  const int chunk = tid & 7;  // 0..7  : 16-element column chunk

  v8f accq[4] = {};
  v8f acck[4] = {};

  for (int cb = 0; cb < CC; cb += 32) {
    // ---- Stage X[cb..cb+31][lbase..lbase+127] into sX ----
#if USE_TDM
    if (tid < 32) {  // one wave issues the DMA (EXEC ignored by TDM)
      const bf16_t* gsrc = xbf + ((size_t)(b * CC + cb)) * LL + lbase;
      const unsigned long long ga = (unsigned long long)(size_t)gsrc;
      // D# group 0: count=1 | lds_addr | global_addr[56:0] | type=2
      u32x4 g0;
      g0[0] = 1u;                                        // count=1 (valid user D#)
      g0[1] = 0u;                                        // lds_addr: sX @ LDS offset 0
      g0[2] = (unsigned int)(ga & 0xFFFFFFFFull);        // global_addr[31:0]
      g0[3] = (unsigned int)((ga >> 32) & 0x01FFFFFFull) // global_addr[56:32]
              | (2u << 30);                              // type = 2 ("image")
      // D# group 1: 2D tile, data_size=2B
      i32x8 g1;
      g1[0] = (1 << 16);                  // [17:16] data_size = 1 -> 2 bytes
      g1[1] = (int)((LL & 0xFFFF) << 16); // [63:48]  tensor_dim0 lo16 = 2048
      g1[2] = (int)((CC & 0xFFFF) << 16); // [79:64] dim0 hi=0 | [95:80] tensor_dim1 lo16 = 512
      g1[3] = (int)(128 << 16);           // [111:96] dim1 hi=0 | [127:112] tile_dim0 = 128
      g1[4] = 32;                         // [143:128] tile_dim1 = 32 | tile_dim2 = 0
      g1[5] = LL;                         // [191:160] tensor_dim0_stride lo32 = 2048
      g1[6] = 0;                          // stride hi | tensor_dim1_stride lo (unused, 2D)
      g1[7] = 0;
      i32x4 z4 = {0, 0, 0, 0};
#if TDM_SIX_ARG
      i32x8 z8 = {0, 0, 0, 0, 0, 0, 0, 0};
      __builtin_amdgcn_tensor_load_to_lds(g0, g1, z4, z4, z8, 0);
#else
      __builtin_amdgcn_tensor_load_to_lds(g0, g1, z4, z4, 0);
#endif
      __builtin_amdgcn_s_wait_tensorcnt(0);
    }
#else
    {
      const bf16_t* src = xbf + ((size_t)(b * CC + cb + row)) * LL + lbase + chunk * 16;
      bf16_t* dst = &sX[row * 128 + chunk * 16];
#if USE_ASYNC_LDS
      __builtin_amdgcn_global_load_async_to_lds_b128((v4i_vs*)src, (v4i_vs*)dst, 0, 0);
      __builtin_amdgcn_global_load_async_to_lds_b128((v4i_vs*)(src + 8), (v4i_vs*)(dst + 8), 0, 0);
      __builtin_amdgcn_s_wait_asynccnt(0);
#else
      v8bf d0 = *(const v8bf*)(src);
      v8bf d1 = *(const v8bf*)(src + 8);
      *(v8bf*)(dst) = d0;
      *(v8bf*)(dst + 8) = d1;
#endif
    }
#endif
    __syncthreads();

    // Preload all A fragments, then a back-to-back WMMA burst (amortizes the
    // WMMA->VALU WAR hazard NOPs over 8 WMMAs instead of paying them per-op).
    v16bf a1[4], a2[4];
#pragma unroll
    for (int mt = 0; mt < 4; ++mt) {
      a1[mt] = load_a_f32g(W1 + (size_t)(mt * 16) * CC + cb, CC, lane);
      a2[mt] = load_a_f32g(W2 + (size_t)(mt * 16) * CC + cb, CC, lane);
    }
    v16bf bx = load_b_lds_bf16(sX, wave * 16, lane);
#pragma unroll
    for (int mt = 0; mt < 4; ++mt) {
      accq[mt] = wmma_bf16(a1[mt], bx, accq[mt]);
      acck[mt] = wmma_bf16(a2[mt], bx, acck[mt]);
    }
    __syncthreads();
  }

  // Store transposed: qT[b][l][t]. D layout: col n = lane&15 (= l), row m = 8*half + r (= t)
  const int n = lane & 15, half = lane >> 4;
  const int lglob = lbase + wave * 16 + n;
  bf16_t* qrow = qT + ((size_t)(b * LL + lglob)) * TT;
  bf16_t* krow = kT + ((size_t)(b * LL + lglob)) * TT;
#pragma unroll
  for (int mt = 0; mt < 4; ++mt) {
    v8bf pq, pk;
#pragma unroll
    for (int r = 0; r < 8; ++r) {
      pq[r] = (__bf16)accq[mt][r];
      pk[r] = (__bf16)acck[mt][r];
    }
    *(v8bf*)(qrow + mt * 16 + half * 8) = pq;
    *(v8bf*)(krow + mt * 16 + half * 8) = pk;
  }
}

// ---------------------------------------------------------------------------
// Kernel 2 (fused): per (batch, 32-row j-strip):
//   scores strip (32 x 2048) = qT_strip x kT^T   -> LDS (f32, 256 KB)
//   row softmax in LDS (wave32 shuffle reductions)
//   out strip (512 x 32) = Xbf (512x2048) x Attn_strip^T   (K = 2048)
//   epilogue: out = x + gamma * out
// Block = 256 threads (8 waves). Dynamic LDS = 256 KB (<= 320 KB/WGP).
// ---------------------------------------------------------------------------
__global__ void __launch_bounds__(256) attn_kernel(const float* __restrict__ x,
                                                   const bf16_t* __restrict__ xbf,
                                                   const bf16_t* __restrict__ qT,
                                                   const bf16_t* __restrict__ kT,
                                                   const float* __restrict__ gamma,
                                                   float* __restrict__ out) {
  extern __shared__ float sS[];  // [32][LL]
  const int b = blockIdx.x >> 6;
  const int jbase = (blockIdx.x & 63) * 32;
  const int tid = threadIdx.x, lane = tid & 31, wave = tid >> 5;
  const int n = lane & 15, half = lane >> 4;

  // ---- Phase A: scores strip = Q_strip^T x K  (M=32, K=T=64, N=2048) ----
  v16bf aq[2][2];
#pragma unroll
  for (int mt = 0; mt < 2; ++mt)
#pragma unroll
    for (int ks = 0; ks < 2; ++ks)
      aq[mt][ks] =
          load_a_bf16g(qT + ((size_t)(b * LL + jbase + mt * 16)) * TT + ks * 32, TT, lane);

  for (int nt = 0; nt < 16; ++nt) {
    const int ncol = (wave * 16 + nt) * 16;  // 8 waves x 16 tiles cover 2048 cols
    v8f s0 = {}, s1 = {};
#pragma unroll
    for (int ks = 0; ks < 2; ++ks) {
      v16bf bk = load_b_bf16g(kT + ((size_t)(b * LL + ncol)) * TT + ks * 32, TT, lane);
      s0 = wmma_bf16(aq[0][ks], bk, s0);
      s1 = wmma_bf16(aq[1][ks], bk, s1);
    }
#pragma unroll
    for (int r = 0; r < 8; ++r) {
      sS[(half * 8 + r) * LL + ncol + n] = s0[r];
      sS[(16 + half * 8 + r) * LL + ncol + n] = s1[r];
    }
  }
  __syncthreads();

  // ---- Phase B: row-wise softmax over 2048 (each wave owns 4 rows) ----
#pragma unroll
  for (int i = 0; i < 4; ++i) {
    float* rowp = sS + (size_t)(wave * 4 + i) * LL;
    float mx = -3.402823466e38f;
    for (int l = lane; l < LL; l += 32) mx = fmaxf(mx, rowp[l]);
#pragma unroll
    for (int off = 16; off > 0; off >>= 1) mx = fmaxf(mx, __shfl_xor(mx, off, 32));
    float sum = 0.f;
    for (int l = lane; l < LL; l += 32) {
      float e = __expf(rowp[l] - mx);
      rowp[l] = e;
      sum += e;
    }
#pragma unroll
    for (int off = 16; off > 0; off >>= 1) sum += __shfl_xor(sum, off, 32);
    const float inv = 1.0f / sum;
    for (int l = lane; l < LL; l += 32) rowp[l] *= inv;
  }
  __syncthreads();

  // ---- Phase C: out strip = Xbf x Attn^T  (M=512: 4 Mtiles/wave, N=32, K=2048) ----
  v8f acc[4][2] = {};
  for (int kb = 0; kb < LL; kb += 32) {
    // Preload operands first, then issue 8 WMMAs back-to-back.
    v16bf ax[4];
#pragma unroll
    for (int m4 = 0; m4 < 4; ++m4) {
      const int c0 = (wave * 4 + m4) * 16;
      const bf16_t* ap = xbf + ((size_t)(b * CC + c0)) * LL + kb;
      if (kb + 32 < LL) __builtin_prefetch(ap + 32, 0, 0);  // global_prefetch_b8
      ax[m4] = load_a_bf16g(ap, LL, lane);
    }
    v16bf bf0 = load_b_lds_f32(sS, 0, kb, lane);
    v16bf bf1 = load_b_lds_f32(sS, 16, kb, lane);
#pragma unroll
    for (int m4 = 0; m4 < 4; ++m4) {
      acc[m4][0] = wmma_bf16(ax[m4], bf0, acc[m4][0]);
      acc[m4][1] = wmma_bf16(ax[m4], bf1, acc[m4][1]);
    }
  }

  // ---- Phase D: epilogue out = x + gamma * out ----
  const float g = gamma[0];
#pragma unroll
  for (int m4 = 0; m4 < 4; ++m4) {
#pragma unroll
    for (int nt = 0; nt < 2; ++nt) {
      const int j = jbase + nt * 16 + n;
#pragma unroll
      for (int r = 0; r < 8; ++r) {
        const int c = (wave * 4 + m4) * 16 + half * 8 + r;
        const size_t idx = ((size_t)(b * CC + c)) * LL + j;
        out[idx] = x[idx] + g * acc[m4][nt][r];
      }
    }
  }
}

// ---------------------------------------------------------------------------
extern "C" void kernel_launch(void* const* d_in, const int* in_sizes, int n_in,
                              void* d_out, int out_size, void* d_ws, size_t ws_size,
                              hipStream_t stream) {
  const float* x = (const float*)d_in[0];
  const float* W1 = (const float*)d_in[1];
  const float* W2 = (const float*)d_in[2];
  const float* gamma = (const float*)d_in[3];
  float* out = (float*)d_out;

  char* ws = (char*)d_ws;
  bf16_t* xbf = (bf16_t*)ws;                                           // B*C*L bf16 (16 MB)
  bf16_t* qT = (bf16_t*)(ws + (size_t)BB * CC * LL * sizeof(bf16_t));  // B*L*T bf16 (2 MB)
  bf16_t* kT = qT + (size_t)BB * LL * TT;                              // B*L*T bf16 (2 MB)

  const int nx = BB * CC * LL;
  xcvt_kernel<<<dim3(2048), dim3(256), 0, stream>>>(x, xbf, nx);
  qk_kernel<<<dim3(BB * (LL / 128)), dim3(256), 0, stream>>>(W1, W2, xbf, qT, kT);
  attn_kernel<<<dim3(BB * (LL / 32)), dim3(256), (size_t)32 * LL * sizeof(float), stream>>>(
      x, xbf, qT, kT, gamma, out);
}
